// ResMambaBlock_79783312490996
// MI455X (gfx1250) — matrix operand order repaired and checked by reading, not compiled, not run on hardware
//
#include <hip/hip_runtime.h>
#include <hip/hip_bf16.h>
#include <math.h>

// ---------------- problem constants ----------------
#define L_SEQ     16384            // 32*32*16 spatial tokens per batch
#define BATCH     2
#define T_TOKENS  32768            // BATCH * L_SEQ
#define C_MODEL   96
#define D_INNER   192
#define D_STATE   16
#define NCHUNK    128              // chunks for parallel scan
#define CHUNKLEN  128              // L_SEQ / NCHUNK
#define N_ENT     6144             // BATCH * D_INNER * D_STATE

// ---------------- WMMA types (gfx1250, wave32) ----------------
typedef __attribute__((ext_vector_type(16))) __bf16       v16bf;
typedef __attribute__((ext_vector_type(8)))  float        v8f;
typedef __attribute__((ext_vector_type(4)))  unsigned int u32x4;

union FragBF { v16bf v; unsigned short s[16]; u32x4 q[2]; };

__device__ inline unsigned short f2bf(float f) {
  unsigned int u = __float_as_uint(f);
  u += 0x7FFFu + ((u >> 16) & 1u);           // round-to-nearest-even
  return (unsigned short)(u >> 16);
}
__device__ inline float siluf(float v)     { return v / (1.f + __expf(-v)); }
__device__ inline float softplusf(float v) { return v > 20.f ? v : log1pf(__expf(v)); }
__device__ inline float wave_sum32(float v) {
#pragma unroll
  for (int o = 16; o >= 1; o >>= 1) v += __shfl_xor(v, o, 32);
  return v;
}

// ---------------- weight convert + zero-pad to bf16 ----------------
__global__ void convert_pad_kernel(const float* __restrict__ src, int srcN, int srcK,
                                   unsigned short* __restrict__ dst, int dstN, int dstK) {
  int i = blockIdx.x * 256 + threadIdx.x;
  if (i >= dstN * dstK) return;
  int r = i / dstK, c = i % dstK;
  float v = (r < srcN && c < srcK) ? src[r * srcK + c] : 0.f;
  dst[i] = f2bf(v);
}

// ---------------- instance-norm statistics per (b,c) over L ----------------
__global__ void instnorm_stats_kernel(const float* __restrict__ x,
                                      float* __restrict__ mean, float* __restrict__ rstd) {
  int bc = blockIdx.x;                               // 0..191
  const float* p = x + (size_t)bc * L_SEQ;
  float s = 0.f, s2 = 0.f;
  for (int i = threadIdx.x; i < L_SEQ; i += 256) { float t = p[i]; s += t; s2 += t * t; }
  __shared__ float sh1[256], sh2[256];
  sh1[threadIdx.x] = s; sh2[threadIdx.x] = s2; __syncthreads();
  for (int o = 128; o > 0; o >>= 1) {
    if (threadIdx.x < o) { sh1[threadIdx.x] += sh1[threadIdx.x + o]; sh2[threadIdx.x] += sh2[threadIdx.x + o]; }
    __syncthreads();
  }
  if (threadIdx.x == 0) {
    float mu = sh1[0] * (1.f / L_SEQ);
    float var = sh2[0] * (1.f / L_SEQ) - mu * mu;
    mean[bc] = mu; rstd[bc] = rsqrtf(var + 1e-5f);
  }
}

// -------- normalize + transpose (B,C,L)->(T,96) + LayerNorm over C, bf16 out --------
__global__ void inorm_ln_kernel(const float* __restrict__ x,
                                const float* __restrict__ mean, const float* __restrict__ rstd,
                                const float* __restrict__ lnw, const float* __restrict__ lnb,
                                float* __restrict__ xf, unsigned short* __restrict__ hbf) {
  int wave = threadIdx.x >> 5, lane = threadIdx.x & 31;
  int tok = blockIdx.x * 8 + wave;
  if (tok >= T_TOKENS) return;
  int b = tok / L_SEQ, l = tok % L_SEQ;
  float v[3]; float s = 0.f, s2 = 0.f;
#pragma unroll
  for (int j = 0; j < 3; ++j) {
    int c = lane + 32 * j;
    int bc = b * C_MODEL + c;
    float t = (x[(size_t)bc * L_SEQ + l] - mean[bc]) * rstd[bc];
    xf[(size_t)tok * C_MODEL + c] = t;
    v[j] = t; s += t; s2 += t * t;
  }
  s = wave_sum32(s); s2 = wave_sum32(s2);
  float mu = s * (1.f / 96.f);
  float var = s2 * (1.f / 96.f) - mu * mu;
  float r = rsqrtf(var + 1e-5f);
#pragma unroll
  for (int j = 0; j < 3; ++j) {
    int c = lane + 32 * j;
    hbf[(size_t)tok * C_MODEL + c] = f2bf((v[j] - mu) * r * lnw[c] + lnb[c]);
  }
}

// -------- LayerNorm over 96 for f32 rows -> bf16 --------
__global__ void ln_rows_kernel(const float* __restrict__ M,
                               const float* __restrict__ g, const float* __restrict__ bta,
                               unsigned short* __restrict__ outbf) {
  int wave = threadIdx.x >> 5, lane = threadIdx.x & 31;
  int tok = blockIdx.x * 8 + wave;
  if (tok >= T_TOKENS) return;
  const float* row = M + (size_t)tok * 96;
  float v[3]; float s = 0.f, s2 = 0.f;
#pragma unroll
  for (int j = 0; j < 3; ++j) { float t = row[lane + 32 * j]; v[j] = t; s += t; s2 += t * t; }
  s = wave_sum32(s); s2 = wave_sum32(s2);
  float mu = s * (1.f / 96.f);
  float var = s2 * (1.f / 96.f) - mu * mu;
  float r = rsqrtf(var + 1e-5f);
#pragma unroll
  for (int j = 0; j < 3; ++j) {
    int c = lane + 32 * j;
    outbf[(size_t)tok * 96 + c] = f2bf((v[j] - mu) * r * g[c] + bta[c]);
  }
}

// ---------------- generic WMMA GEMM: Y(MxN) = X(MxK,bf16) * W^T(NxK,bf16) ----------------
// One wave per 16x16 tile, K-loop of 32 via v_wmma_f32_16x16x32_bf16.
// epi: 0=store f32, 1=+bias,softplus (delta), 2=+skip*aux (out_proj), 3=+bias,+identity,relu,
//      transposed store to (B,C,L)   (final proj writes d_out directly)
__global__ void wmma_gemm_kernel(const unsigned short* __restrict__ X, int ldx,
                                 const unsigned short* __restrict__ W, int K,
                                 float* __restrict__ Y, int ldy,
                                 int M, int N, int epi,
                                 const float* __restrict__ bias,
                                 const float* __restrict__ aux,       // xf for epi 2
                                 const float* __restrict__ skipscale, // [1]  for epi 2
                                 const float* __restrict__ ident) {   // x    for epi 3
  const int wave = threadIdx.x >> 5;
  const int lane = threadIdx.x & 31;
  const int ntiles = N >> 4;
  const int tile = blockIdx.x * (blockDim.x >> 5) + wave;
  const int mt = tile / ntiles;
  const int nt = tile % ntiles;
  if (mt * 16 >= M) return;                          // wave-uniform; EXEC stays all-1s

  // A fragment addressing (16x32 A, same k order as B): lane holds row m, 16 contiguous K halfs.
  const int mrow  = mt * 16 + (lane & 15);
  const int khalf = (lane >> 4) * 8;
  // B fragment addressing (32x16 B, k = 16*(lane/16) + s, same as A): lane holds col n, 16 contiguous K halfs.
  const int ncol = nt * 16 + (lane & 15);
  const int kb   = (lane >> 4) * 16;

  const unsigned short* pa = X + (size_t)mrow * ldx;
  const unsigned short* pb = W + (size_t)ncol * K;

  v8f acc = {0.f, 0.f, 0.f, 0.f, 0.f, 0.f, 0.f, 0.f};
  for (int k0 = 0; k0 < K; k0 += 32) {
    FragBF a, b;
    a.q[0] = *reinterpret_cast<const u32x4*>(pa + k0 + kb);          // K kb..kb+7
    a.q[1] = *reinterpret_cast<const u32x4*>(pa + k0 + kb + 8);      // K kb+8..kb+15
    b.q[0] = *reinterpret_cast<const u32x4*>(pb + k0 + kb);          // K kb..kb+7
    b.q[1] = *reinterpret_cast<const u32x4*>(pb + k0 + kb + 8);      // K kb+8..kb+15
    acc = __builtin_amdgcn_wmma_f32_16x16x32_bf16(false, a.v, false, b.v,
                                                  (short)0, acc, false, false);
  }

  // C/D layout: VGPR r -> M = mt*16 + r + (lane>=16 ? 8 : 0), N = nt*16 + (lane&15)
  const int n = nt * 16 + (lane & 15);
#pragma unroll
  for (int r = 0; r < 8; ++r) {
    int mm = mt * 16 + r + ((lane >> 4) << 3);
    float v = acc[r];
    if (bias) v += bias[n];
    if (epi == 1) {
      Y[(size_t)mm * ldy + n] = softplusf(v);
    } else if (epi == 2) {
      Y[(size_t)mm * ldy + n] = v + skipscale[0] * aux[(size_t)mm * ldy + n];
    } else if (epi == 3) {
      int bb = mm / L_SEQ, ll = mm % L_SEQ;
      size_t oi = ((size_t)bb * C_MODEL + n) * L_SEQ + ll;
      float o = v + ident[oi];
      Y[oi] = o > 0.f ? o : 0.f;
    } else {
      Y[(size_t)mm * ldy + n] = v;
    }
  }
}

// ---------------- depthwise causal conv (taps=4) + SiLU ----------------
__global__ void conv_silu_kernel(const float* __restrict__ xz,
                                 const float* __restrict__ cw, const float* __restrict__ cb,
                                 float* __restrict__ uf, unsigned short* __restrict__ ubf) {
  int i = blockIdx.x * 256 + threadIdx.x;            // over T_TOKENS * D_INNER
  int d = i % D_INNER, tok = i / D_INNER;
  int l = tok % L_SEQ, b = tok / L_SEQ;
  float s = cb[d];
#pragma unroll
  for (int t = 0; t < 4; ++t) {
    int ll = l - 3 + t;
    if (ll >= 0) s += cw[d * 4 + t] * xz[((size_t)b * L_SEQ + ll) * 384 + d];
  }
  float v = siluf(s);
  uf[i] = v; ubf[i] = f2bf(v);
}

// ---------------- gather dt_r (first 6 of xdbl) into K=32 zero-padded bf16 ----------------
__global__ void prep_dtr_kernel(const float* __restrict__ xdbl, unsigned short* __restrict__ dtrbf) {
  int i = blockIdx.x * 256 + threadIdx.x;            // over T_TOKENS * 32
  int c = i & 31, tok = i >> 5;
  float v = (c < 6) ? xdbl[(size_t)tok * 48 + c] : 0.f;
  dtrbf[i] = f2bf(v);
}

// ---------------- chunked parallel scan ----------------
// entity = b*3072 + d*16 + n ; recurrence h = a*h + bu with a = exp(delta*A[d,n]),
// bu = delta * Bm[n] * u[d].  Pass1: per (entity,chunk) compute (P=prod a, S=affine offset).
__global__ void scan_pass1(const float* __restrict__ delta, const float* __restrict__ u,
                           const float* __restrict__ xdbl, const float* __restrict__ A_log,
                           float* __restrict__ P, float* __restrict__ S) {
  int t = blockIdx.x * 256 + threadIdx.x;            // NCHUNK * N_ENT threads
  int ent = t % N_ENT, chunk = t / N_ENT;
  int n = ent & 15, d = (ent >> 4) % D_INNER, b = ent / (D_INNER * D_STATE);
  float A = -__expf(A_log[d * 16 + n]);
  const float* del = delta + (size_t)b * L_SEQ * D_INNER + d;
  const float* uu  = u     + (size_t)b * L_SEQ * D_INNER + d;
  const float* bm  = xdbl  + (size_t)b * L_SEQ * 48 + 6 + n;
  float p = 1.f, s = 0.f;
  int l0 = chunk * CHUNKLEN;
  for (int i = 0; i < CHUNKLEN; ++i) {
    int l = l0 + i;
    if ((i & 15) == 0 && l + 32 < L_SEQ)
      __builtin_prefetch(del + (size_t)(l + 32) * D_INNER, 0, 0);   // global_prefetch
    float dlt = del[(size_t)l * D_INNER];
    float a = __expf(dlt * A);
    float bu = dlt * bm[(size_t)l * 48] * uu[(size_t)l * D_INNER];
    p *= a;
    s = a * s + bu;
  }
  P[t] = p; S[t] = s;
}

// Pass2: sequential composition across 128 chunks per entity (depth-128 only).
__global__ void scan_pass2(const float* __restrict__ P, const float* __restrict__ S,
                           float* __restrict__ H) {
  int ent = blockIdx.x * 256 + threadIdx.x;          // N_ENT threads
  float h = 0.f;
  for (int c = 0; c < NCHUNK; ++c) {
    size_t i = (size_t)c * N_ENT + ent;
    H[i] = h;                                        // initial state for chunk c
    h = P[i] * h + S[i];
  }
}

// Pass3: replay chunks in parallel with correct initial state; reduce over n via shfl.
__global__ void scan_pass3(const float* __restrict__ delta, const float* __restrict__ u,
                           const float* __restrict__ xdbl, const float* __restrict__ A_log,
                           const float* __restrict__ H, float* __restrict__ ys) {
  int t = blockIdx.x * 256 + threadIdx.x;
  int ent = t % N_ENT, chunk = t / N_ENT;
  int n = ent & 15, d = (ent >> 4) % D_INNER, b = ent / (D_INNER * D_STATE);
  float A = -__expf(A_log[d * 16 + n]);
  const float* del = delta + (size_t)b * L_SEQ * D_INNER + d;
  const float* uu  = u     + (size_t)b * L_SEQ * D_INNER + d;
  const float* bm  = xdbl  + (size_t)b * L_SEQ * 48 + 6 + n;
  const float* cm  = xdbl  + (size_t)b * L_SEQ * 48 + 22 + n;
  float* yo = ys + (size_t)b * L_SEQ * D_INNER + d;
  float h = H[(size_t)chunk * N_ENT + ent];
  int l0 = chunk * CHUNKLEN;
  for (int i = 0; i < CHUNKLEN; ++i) {
    int l = l0 + i;
    float dlt = del[(size_t)l * D_INNER];
    float uv  = uu[(size_t)l * D_INNER];
    float a = __expf(dlt * A);
    h = a * h + dlt * bm[(size_t)l * 48] * uv;
    float yv = h * cm[(size_t)l * 48];
#pragma unroll
    for (int o = 8; o >= 1; o >>= 1) yv += __shfl_xor(yv, o, 32);   // sum over 16 states
    if (n == 0) yo[(size_t)l * D_INNER] = yv;
  }
}

// ---------------- y = (ys + u*D) * silu(z) -> bf16 ----------------
__global__ void gate_kernel(const float* __restrict__ ys, const float* __restrict__ uf,
                            const float* __restrict__ xz, const float* __restrict__ Dv,
                            unsigned short* __restrict__ ybf) {
  int i = blockIdx.x * 256 + threadIdx.x;            // T_TOKENS * D_INNER
  int d = i % D_INNER, tok = i / D_INNER;
  float yv = ys[i] + uf[i] * Dv[d];
  float z = xz[(size_t)tok * 384 + 192 + d];
  ybf[i] = f2bf(yv * siluf(z));
}

// ================================ host launch ================================
extern "C" void kernel_launch(void* const* d_in, const int* in_sizes, int n_in,
                              void* d_out, int out_size, void* d_ws, size_t ws_size,
                              hipStream_t stream) {
  (void)in_sizes; (void)n_in; (void)out_size; (void)ws_size;
  const float* x         = (const float*)d_in[0];
  const float* in_proj_w = (const float*)d_in[1];
  const float* conv_w    = (const float*)d_in[2];
  const float* conv_b    = (const float*)d_in[3];
  const float* x_proj_w  = (const float*)d_in[4];
  const float* dt_proj_w = (const float*)d_in[5];
  const float* dt_proj_b = (const float*)d_in[6];
  const float* A_log     = (const float*)d_in[7];
  const float* Dvec      = (const float*)d_in[8];
  const float* out_proj_w= (const float*)d_in[9];
  const float* ln_w      = (const float*)d_in[10];
  const float* ln_b      = (const float*)d_in[11];
  const float* proj_w    = (const float*)d_in[12];
  const float* proj_b    = (const float*)d_in[13];
  const float* skip      = (const float*)d_in[14];
  float* out = (float*)d_out;

  char* wp = (char*)d_ws;
  auto alloc = [&](size_t bytes) -> char* {
    char* p = wp; wp += (bytes + 255) & ~(size_t)255; return p;
  };
  float* mean           = (float*)alloc(192 * 4);
  float* rstd           = (float*)alloc(192 * 4);
  float* xf             = (float*)alloc((size_t)T_TOKENS * 96 * 4);
  unsigned short* hbf   = (unsigned short*)alloc((size_t)T_TOKENS * 96 * 2);
  float* xz             = (float*)alloc((size_t)T_TOKENS * 384 * 4);
  float* uf             = (float*)alloc((size_t)T_TOKENS * 192 * 4);
  unsigned short* ubf   = (unsigned short*)alloc((size_t)T_TOKENS * 192 * 2);
  float* xdbl           = (float*)alloc((size_t)T_TOKENS * 48 * 4);
  unsigned short* dtrbf = (unsigned short*)alloc((size_t)T_TOKENS * 32 * 2);
  float* delta          = (float*)alloc((size_t)T_TOKENS * 192 * 4);
  float* Pc             = (float*)alloc((size_t)NCHUNK * N_ENT * 4);
  float* Sc             = (float*)alloc((size_t)NCHUNK * N_ENT * 4);
  float* Hc             = (float*)alloc((size_t)NCHUNK * N_ENT * 4);
  float* ys             = (float*)alloc((size_t)T_TOKENS * 192 * 4);
  unsigned short* ybf   = (unsigned short*)alloc((size_t)T_TOKENS * 192 * 2);
  float* mbuf           = (float*)alloc((size_t)T_TOKENS * 96 * 4);
  unsigned short* mnbf  = (unsigned short*)alloc((size_t)T_TOKENS * 96 * 2);
  unsigned short* w_in  = (unsigned short*)alloc(384 * 96 * 2);
  unsigned short* w_xp  = (unsigned short*)alloc(48 * 192 * 2);
  unsigned short* w_dt  = (unsigned short*)alloc(192 * 32 * 2);
  unsigned short* w_out = (unsigned short*)alloc(96 * 192 * 2);
  unsigned short* w_pr  = (unsigned short*)alloc(96 * 96 * 2);

  // weights -> bf16 (pad x_proj N 38->48, dt_proj K 6->32 with zeros)
  convert_pad_kernel<<<(384 * 96 + 255) / 256, 256, 0, stream>>>(in_proj_w, 384, 96, w_in, 384, 96);
  convert_pad_kernel<<<(48 * 192 + 255) / 256, 256, 0, stream>>>(x_proj_w, 38, 192, w_xp, 48, 192);
  convert_pad_kernel<<<(192 * 32 + 255) / 256, 256, 0, stream>>>(dt_proj_w, 192, 6, w_dt, 192, 32);
  convert_pad_kernel<<<(96 * 192 + 255) / 256, 256, 0, stream>>>(out_proj_w, 96, 192, w_out, 96, 192);
  convert_pad_kernel<<<(96 * 96 + 255) / 256, 256, 0, stream>>>(proj_w, 96, 96, w_pr, 96, 96);

  // instance norm + transpose + LN -> xf (f32 skip path), hbf (bf16 GEMM operand)
  instnorm_stats_kernel<<<192, 256, 0, stream>>>(x, mean, rstd);
  inorm_ln_kernel<<<T_TOKENS / 8, 256, 0, stream>>>(x, mean, rstd, ln_w, ln_b, xf, hbf);

  // in_proj: (32768x96)*(384x96)^T -> xz
  wmma_gemm_kernel<<<(T_TOKENS / 16) * (384 / 16) / 8, 256, 0, stream>>>(
      hbf, 96, w_in, 96, xz, 384, T_TOKENS, 384, 0, nullptr, nullptr, nullptr, nullptr);

  // depthwise conv + SiLU -> u
  conv_silu_kernel<<<(T_TOKENS * 192) / 256, 256, 0, stream>>>(xz, conv_w, conv_b, uf, ubf);

  // x_proj: (32768x192)*(48x192)^T -> xdbl (dt_r | Bm | Cm, padded)
  wmma_gemm_kernel<<<(T_TOKENS / 16) * (48 / 16) / 8, 256, 0, stream>>>(
      ubf, 192, w_xp, 192, xdbl, 48, T_TOKENS, 48, 0, nullptr, nullptr, nullptr, nullptr);

  // dt_proj + softplus -> delta
  prep_dtr_kernel<<<(T_TOKENS * 32) / 256, 256, 0, stream>>>(xdbl, dtrbf);
  wmma_gemm_kernel<<<(T_TOKENS / 16) * (192 / 16) / 8, 256, 0, stream>>>(
      dtrbf, 32, w_dt, 32, delta, 192, T_TOKENS, 192, 1, dt_proj_b, nullptr, nullptr, nullptr);

  // chunked parallel selective scan (depth 128 instead of 16384)
  scan_pass1<<<(NCHUNK * N_ENT) / 256, 256, 0, stream>>>(delta, uf, xdbl, A_log, Pc, Sc);
  scan_pass2<<<N_ENT / 256, 256, 0, stream>>>(Pc, Sc, Hc);
  scan_pass3<<<(NCHUNK * N_ENT) / 256, 256, 0, stream>>>(delta, uf, xdbl, A_log, Hc, ys);

  // gate: (ys + u*D) * silu(z) -> bf16
  gate_kernel<<<(T_TOKENS * 192) / 256, 256, 0, stream>>>(ys, uf, xz, Dvec, ybf);

  // out_proj + skip*xf -> m
  wmma_gemm_kernel<<<(T_TOKENS / 16) * (96 / 16) / 8, 256, 0, stream>>>(
      ybf, 192, w_out, 192, mbuf, 96, T_TOKENS, 96, 2, nullptr, xf, skip, nullptr);

  // second LayerNorm -> bf16
  ln_rows_kernel<<<T_TOKENS / 8, 256, 0, stream>>>(mbuf, ln_w, ln_b, mnbf);

  // final proj + bias + residual + relu, transposed store straight to d_out
  wmma_gemm_kernel<<<(T_TOKENS / 16) * (96 / 16) / 8, 256, 0, stream>>>(
      mnbf, 96, w_pr, 96, out, 96, T_TOKENS, 96, 3, proj_b, nullptr, nullptr, x);
}
